// longformer_Block_14843406975618
// MI455X (gfx1250) — matrix operand order repaired and checked by reading, not compiled, not run on hardware
//
#include <hip/hip_runtime.h>
#include <stdint.h>

// ---- problem constants (match reference) ----
#define B_   2
#define S_   4096
#define D_   1024
#define H_   16
#define F_   4096
#define W_   256
#define DH_  64
#define NB_  (S_ / W_)     // 16 blocks
#define MTOT (B_ * S_)     // 8192 rows

// ---- CDNA5 WMMA types ----
typedef __attribute__((ext_vector_type(16))) __bf16        v16bf;
typedef __attribute__((ext_vector_type(8)))  float         v8f;
typedef __attribute__((ext_vector_type(4)))  unsigned int  v4u;
typedef __attribute__((ext_vector_type(4)))  int           v4i;

union Frag16 {
  v16bf          bf;
  v4u            q[2];
  unsigned short us[16];
};

__device__ __forceinline__ unsigned short f32_to_bf16(float f) {
  union { float f; unsigned u; } v; v.f = f;
  unsigned r = v.u + 0x7FFFu + ((v.u >> 16) & 1u);  // round-to-nearest-even
  return (unsigned short)(r >> 16);
}
__device__ __forceinline__ float bf16_to_f32(unsigned short h) {
  union { unsigned u; float f; } v; v.u = ((unsigned)h) << 16;
  return v.f;
}

// ---- gfx1250 async global->LDS copy (ASYNCcnt path), with safe fallback ----
#define AS3 __attribute__((address_space(3)))

__device__ __forceinline__ void async_copy16(unsigned short* l, const unsigned short* g) {
#if __has_builtin(__builtin_amdgcn_global_load_async_to_lds_b128)
  __builtin_amdgcn_global_load_async_to_lds_b128((v4i*)g, (AS3 v4i*)l, 0, 0);
#else
  *(v4u*)l = *(const v4u*)g;
#endif
}
__device__ __forceinline__ void wait_async_all() {
#if __has_builtin(__builtin_amdgcn_s_wait_asynccnt)
  __builtin_amdgcn_s_wait_asynccnt(0);
#endif
}

// =====================================================================
// Block-tiled GEMM core: workgroup (8 waves) computes 128(M) x 128(N).
// Per 32-wide k-slice, A(128x32) and B(128x32) tiles are staged into a
// double-buffered LDS (async DMA overlapped with WMMA on the previous
// slice). Each wave computes 32x64: 2 A-fragments x 4 B-fragments -> 8
// WMMAs per slice (16 for dual-B). Global traffic: 16KB per 1 MFLOP.
// Fragment layouts (contiguous k order, same for A and B):
//   A (16x32): lane half 0 holds K {k..k+15}, half 1 holds
//              K {k+16..k+31} for row (lane&15).
//   B (32x16): lane = column n, 16 contiguous K starting at 0 (+16 for
//              lane half 1). Staged rows are [n][k] so this is 32B.
// =====================================================================
template <int NBMAT>
__device__ __forceinline__ void gemm_block_core(
    const unsigned short* __restrict__ A,
    const unsigned short* __restrict__ B0t,
    const unsigned short* __restrict__ B1t,
    int K, int m0blk, int n0blk,
    unsigned short* sA, unsigned short* sB0, unsigned short* sB1,
    v8f (&acc)[NBMAT][2][4]) {
  const int tid  = threadIdx.x;
  const int lane = tid & 31, wid = tid >> 5, half = lane >> 4, lr = lane & 15;
  const int wm = wid >> 1, wn = wid & 1;           // 4 x 2 wave grid

  // stage one 128x32 bf16 tile: 512 16B-chunks, 2 per thread
  const int c0  = tid * 2;
  const int r0  = c0 >> 2,  kh0 = (c0 & 3) * 8;
  const int r1  = (c0 + 1) >> 2, kh1 = ((c0 + 1) & 3) * 8;

  auto stage = [&](unsigned short* sbuf, const unsigned short* gbase, int buf, int k) {
    const unsigned short* g0 = gbase + (size_t)r0 * K + k + kh0;
    const unsigned short* g1 = gbase + (size_t)r1 * K + k + kh1;
    unsigned short* l = sbuf + buf * (128 * 32);
    async_copy16(l + r0 * 32 + kh0, g0);
    async_copy16(l + r1 * 32 + kh1, g1);
  };

  const unsigned short* Ab  = A   + (size_t)m0blk * K;
  const unsigned short* B0b = B0t + (size_t)n0blk * K;
  const unsigned short* B1b = B1t + (size_t)n0blk * K;

  stage(sA, Ab, 0, 0);
  stage(sB0, B0b, 0, 0);
  if constexpr (NBMAT == 2) stage(sB1, B1b, 0, 0);

  const int nk = K / 32;
  for (int i = 0; i < nk; ++i) {
    const int cur = i & 1;
    wait_async_all();
    __syncthreads();                       // tile i resident for all waves
    if (i + 1 < nk) {                      // prefetch tile i+1 into other buffer
      stage(sA, Ab, cur ^ 1, (i + 1) * 32);
      stage(sB0, B0b, cur ^ 1, (i + 1) * 32);
      if constexpr (NBMAT == 2) stage(sB1, B1b, cur ^ 1, (i + 1) * 32);
    }
    // fragments from LDS
    const unsigned short* a0 = sA + cur * (128 * 32) + (wm * 32 + lr) * 32;
    const unsigned short* a1 = sA + cur * (128 * 32) + (wm * 32 + 16 + lr) * 32;
    const int ka = half * 16;
    Frag16 fa0, fa1;
    fa0.q[0] = *(const v4u*)(a0 + ka); fa0.q[1] = *(const v4u*)(a0 + ka + 8);
    fa1.q[0] = *(const v4u*)(a1 + ka); fa1.q[1] = *(const v4u*)(a1 + ka + 8);
#pragma unroll
    for (int t = 0; t < 4; ++t) {
      const int brow = (wn * 64 + t * 16 + lr) * 32 + half * 16;
      Frag16 fb0;
      const unsigned short* bp0 = sB0 + cur * (128 * 32) + brow;
      fb0.q[0] = *(const v4u*)(bp0); fb0.q[1] = *(const v4u*)(bp0 + 8);
      acc[0][0][t] = __builtin_amdgcn_wmma_f32_16x16x32_bf16(false, fa0.bf, false, fb0.bf,
                                                             (short)0, acc[0][0][t], false, false);
      acc[0][1][t] = __builtin_amdgcn_wmma_f32_16x16x32_bf16(false, fa1.bf, false, fb0.bf,
                                                             (short)0, acc[0][1][t], false, false);
      if constexpr (NBMAT == 2) {
        Frag16 fb1;
        const unsigned short* bp1 = sB1 + cur * (128 * 32) + brow;
        fb1.q[0] = *(const v4u*)(bp1); fb1.q[1] = *(const v4u*)(bp1 + 8);
        acc[1][0][t] = __builtin_amdgcn_wmma_f32_16x16x32_bf16(false, fa0.bf, false, fb1.bf,
                                                               (short)0, acc[1][0][t], false, false);
        acc[1][1][t] = __builtin_amdgcn_wmma_f32_16x16x32_bf16(false, fa1.bf, false, fb1.bf,
                                                               (short)0, acc[1][1][t], false, false);
      }
    }
    __syncthreads();                       // done reading before buffer reuse
  }
}

// ---- weight convert + transpose: W[K][N] f32 -> Wt[N][K] bf16 ----
__global__ void convt_kernel(const float* __restrict__ w, unsigned short* __restrict__ wt,
                             int K, int N) {
  long long idx = (long long)blockIdx.x * 256 + threadIdx.x;
  if (idx >= (long long)K * N) return;
  int k = (int)(idx / N), n = (int)(idx % N);
  wt[(size_t)n * K + k] = f32_to_bf16(w[idx]);
}

// ---- RMSNorm: row-per-block, fp32 in -> bf16 out ----
__global__ void __launch_bounds__(256)
rmsnorm_kernel(const float* __restrict__ x, const float* __restrict__ g,
               unsigned short* __restrict__ out) {
  __shared__ float red[256];
  const int row = blockIdx.x;
  const float* xr = x + (size_t)row * D_;
  float v[4];
  float p = 0.0f;
#pragma unroll
  for (int i = 0; i < 4; ++i) { v[i] = xr[threadIdx.x * 4 + i]; p += v[i] * v[i]; }
  red[threadIdx.x] = p;
  __syncthreads();
  for (int s = 128; s > 0; s >>= 1) {
    if ((int)threadIdx.x < s) red[threadIdx.x] += red[threadIdx.x + s];
    __syncthreads();
  }
  const float r = rsqrtf(red[0] / (float)D_ + 1e-6f);
#pragma unroll
  for (int i = 0; i < 4; ++i) {
    int c = threadIdx.x * 4 + i;
    out[(size_t)row * D_ + c] = f32_to_bf16(v[i] * r * g[c]);
  }
}

// ---- GEMM + bias -> bf16 (row-major [M][N]) ----
__global__ void __launch_bounds__(256)
gemm_bias_bf16(const unsigned short* __restrict__ A, const unsigned short* __restrict__ Bt,
               const float* __restrict__ bias, unsigned short* __restrict__ C,
               int M, int N, int K) {
  __shared__ unsigned short sA[2 * 128 * 32];
  __shared__ unsigned short sB[2 * 128 * 32];
  const int lane = threadIdx.x & 31, wid = threadIdx.x >> 5, half = lane >> 4, lr = lane & 15;
  const int nb = N / 128;
  const int m0blk = (blockIdx.x / nb) * 128;
  const int n0blk = (blockIdx.x % nb) * 128;
  const int m0 = m0blk + (wid >> 1) * 32;
  const int n0 = n0blk + (wid & 1) * 64;
  v8f acc[1][2][4] = {};
  gemm_block_core<1>(A, Bt, Bt, K, m0blk, n0blk, sA, sB, sB, acc);
#pragma unroll
  for (int mi = 0; mi < 2; ++mi)
#pragma unroll
    for (int t = 0; t < 4; ++t)
#pragma unroll
      for (int r = 0; r < 8; ++r) {
        int n = n0 + t * 16 + lr;
        int m = m0 + mi * 16 + half * 8 + r;
        C[(size_t)m * N + n] = f32_to_bf16(acc[0][mi][t][r] + bias[n]);
      }
}

// ---- GEMM + bias -> bf16 V stored head-transposed: vt[b][h][dh][s] ----
__global__ void __launch_bounds__(256)
gemm_bias_vt(const unsigned short* __restrict__ A, const unsigned short* __restrict__ Bt,
             const float* __restrict__ bias, unsigned short* __restrict__ vt,
             int M, int N, int K) {
  __shared__ unsigned short sA[2 * 128 * 32];
  __shared__ unsigned short sB[2 * 128 * 32];
  const int lane = threadIdx.x & 31, wid = threadIdx.x >> 5, half = lane >> 4, lr = lane & 15;
  const int nb = N / 128;
  const int m0blk = (blockIdx.x / nb) * 128;
  const int n0blk = (blockIdx.x % nb) * 128;
  const int m0 = m0blk + (wid >> 1) * 32;
  const int n0 = n0blk + (wid & 1) * 64;
  v8f acc[1][2][4] = {};
  gemm_block_core<1>(A, Bt, Bt, K, m0blk, n0blk, sA, sB, sB, acc);
#pragma unroll
  for (int mi = 0; mi < 2; ++mi)
#pragma unroll
    for (int t = 0; t < 4; ++t)
#pragma unroll
      for (int r = 0; r < 8; ++r) {
        int n = n0 + t * 16 + lr;
        int m = m0 + mi * 16 + half * 8 + r;
        int bb = m / S_, s = m % S_;
        int hh = n / DH_, dh = n % DH_;
        vt[(size_t)((bb * H_ + hh) * DH_ + dh) * S_ + s] = f32_to_bf16(acc[0][mi][t][r] + bias[n]);
      }
}

// ---- GEMM + bias + residual -> fp32 ----
__global__ void __launch_bounds__(256)
gemm_bias_resid(const unsigned short* __restrict__ A, const unsigned short* __restrict__ Bt,
                const float* __restrict__ bias, const float* __restrict__ resid,
                float* __restrict__ C, int M, int N, int K) {
  __shared__ unsigned short sA[2 * 128 * 32];
  __shared__ unsigned short sB[2 * 128 * 32];
  const int lane = threadIdx.x & 31, wid = threadIdx.x >> 5, half = lane >> 4, lr = lane & 15;
  const int nb = N / 128;
  const int m0blk = (blockIdx.x / nb) * 128;
  const int n0blk = (blockIdx.x % nb) * 128;
  const int m0 = m0blk + (wid >> 1) * 32;
  const int n0 = n0blk + (wid & 1) * 64;
  v8f acc[1][2][4] = {};
  gemm_block_core<1>(A, Bt, Bt, K, m0blk, n0blk, sA, sB, sB, acc);
#pragma unroll
  for (int mi = 0; mi < 2; ++mi)
#pragma unroll
    for (int t = 0; t < 4; ++t)
#pragma unroll
      for (int r = 0; r < 8; ++r) {
        int n = n0 + t * 16 + lr;
        int m = m0 + mi * 16 + half * 8 + r;
        C[(size_t)m * N + n] = resid[(size_t)m * N + n] + acc[0][mi][t][r] + bias[n];
      }
}

// ---- dual-B SwiGLU GEMM: u = silu(A*w1t + b1) * (A*w3t + b3) -> bf16 ----
__global__ void __launch_bounds__(256)
gemm_swiglu(const unsigned short* __restrict__ A,
            const unsigned short* __restrict__ B1t, const unsigned short* __restrict__ B3t,
            const float* __restrict__ b1, const float* __restrict__ b3,
            unsigned short* __restrict__ U, int M, int N, int K) {
  __shared__ unsigned short sA[2 * 128 * 32];
  __shared__ unsigned short sB1[2 * 128 * 32];
  __shared__ unsigned short sB3[2 * 128 * 32];
  const int lane = threadIdx.x & 31, wid = threadIdx.x >> 5, half = lane >> 4, lr = lane & 15;
  const int nb = N / 128;
  const int m0blk = (blockIdx.x / nb) * 128;
  const int n0blk = (blockIdx.x % nb) * 128;
  const int m0 = m0blk + (wid >> 1) * 32;
  const int n0 = n0blk + (wid & 1) * 64;
  v8f acc[2][2][4] = {};
  gemm_block_core<2>(A, B1t, B3t, K, m0blk, n0blk, sA, sB1, sB3, acc);
#pragma unroll
  for (int mi = 0; mi < 2; ++mi)
#pragma unroll
    for (int t = 0; t < 4; ++t)
#pragma unroll
      for (int r = 0; r < 8; ++r) {
        int n = n0 + t * 16 + lr;
        int m = m0 + mi * 16 + half * 8 + r;
        float a1v = acc[0][mi][t][r] + b1[n];
        float a3v = acc[1][mi][t][r] + b3[n];
        float si = a1v / (1.0f + __expf(-a1v));
        U[(size_t)m * N + n] = f32_to_bf16(si * a3v);
      }
}

// =====================================================================
// Sliding-window attention. One workgroup = (b, h, block, 32-query chunk).
// Phase 1: scores = Q*K^T/8 via WMMA -> bf16 LDS [32][768].
// Phase 2: banded-mask softmax (wave32 shfl reductions), probs in place.
// Phase 3: out = P*V via WMMA (A from LDS, B = vt with OOB chunks zeroed).
// =====================================================================
__global__ void __launch_bounds__(256)
attn_kernel(const unsigned short* __restrict__ qb, const unsigned short* __restrict__ kb,
            const unsigned short* __restrict__ vt, unsigned short* __restrict__ attnb) {
  __shared__ unsigned short sS[32][776];   // 768 + 8 pad, 48.5 KB
  const int tid  = threadIdx.x;
  const int lane = tid & 31, wid = tid >> 5, half = lane >> 4, lr = lane & 15;
  const int bid = blockIdx.x;
  const int qc  = bid & 7;
  const int blk = (bid >> 3) & (NB_ - 1);
  const int h   = (bid >> 7) & (H_ - 1);
  const int b   = bid >> 11;
  const int q0  = blk * W_ + qc * 32;      // first query position of this chunk

  // ---- Phase 1: Q*K^T ----
  for (int tile = wid; tile < 96; tile += 8) {     // 2 Mtiles x 48 Ntiles
    const int mt = tile & 1, nt = tile >> 1;
    const unsigned short* arow = qb + (size_t)(b * S_ + q0 + mt * 16 + lr) * D_ + h * DH_;
    const int j = blk * W_ - W_ + nt * 16 + lr;    // key position for this lane's column
    const bool jok = (j >= 0) && (j < S_);
    const unsigned short* brow = kb + (size_t)(b * S_ + (jok ? j : 0)) * D_ + h * DH_;
    v8f acc = {};
#pragma unroll
    for (int k = 0; k < DH_; k += 32) {
      Frag16 a;
      const int ka = k + half * 16;
      a.q[0] = *(const v4u*)(arow + ka);
      a.q[1] = *(const v4u*)(arow + ka + 8);
      Frag16 bf;
      if (jok) {
        const unsigned short* bp = brow + k + half * 16;
        bf.q[0] = *(const v4u*)(bp);
        bf.q[1] = *(const v4u*)(bp + 8);
      } else {
#pragma unroll
        for (int e = 0; e < 16; ++e) bf.us[e] = 0;
      }
      acc = __builtin_amdgcn_wmma_f32_16x16x32_bf16(false, a.bf, false, bf.bf,
                                                    (short)0, acc, false, false);
    }
#pragma unroll
    for (int r = 0; r < 8; ++r)
      sS[mt * 16 + half * 8 + r][nt * 16 + lr] = f32_to_bf16(acc[r] * 0.125f);
  }
  __syncthreads();

  // ---- Phase 2: masked softmax, one wave per 4 rows ----
  for (int rr = 0; rr < 4; ++rr) {
    const int row = wid * 4 + rr;
    const int qi  = qc * 32 + row;                 // query index within block, 0..255
    float mx = -3.0e38f;
    for (int c = lane; c < 768; c += 32) {
      int j = blk * W_ - W_ + c;
      bool valid = (c >= qi) && (c <= qi + 2 * W_) && (j >= 0) && (j < S_);
      float s = valid ? bf16_to_f32(sS[row][c]) : -3.0e38f;
      mx = fmaxf(mx, s);
    }
#pragma unroll
    for (int off = 16; off > 0; off >>= 1) mx = fmaxf(mx, __shfl_xor(mx, off, 32));
    float sum = 0.0f;
    for (int c = lane; c < 768; c += 32) {
      int j = blk * W_ - W_ + c;
      bool valid = (c >= qi) && (c <= qi + 2 * W_) && (j >= 0) && (j < S_);
      float e = valid ? __expf(bf16_to_f32(sS[row][c]) - mx) : 0.0f;
      sum += e;
      sS[row][c] = f32_to_bf16(e);
    }
#pragma unroll
    for (int off = 16; off > 0; off >>= 1) sum += __shfl_xor(sum, off, 32);
    const float inv = 1.0f / sum;
    for (int c = lane; c < 768; c += 32)
      sS[row][c] = f32_to_bf16(bf16_to_f32(sS[row][c]) * inv);
  }
  __syncthreads();

  // ---- Phase 3: P*V, one 16x16 tile per wave (2 Mtiles x 4 dh-tiles) ----
  {
    const int mt = wid & 1, nt = wid >> 1;
    const unsigned short* arow = &sS[mt * 16 + lr][0];
    const int dh = nt * 16 + lr;
    const unsigned short* vrow = vt + (size_t)((b * H_ + h) * DH_ + dh) * S_;
    v8f acc = {};
    for (int k = 0; k < 768; k += 32) {
      Frag16 a;
      const int ka = k + half * 16;
      a.q[0] = *(const v4u*)(arow + ka);
      a.q[1] = *(const v4u*)(arow + ka + 8);
      const int j0 = blk * W_ - W_ + k + half * 16; // 16-aligned chunk: all-in or all-out
      Frag16 bf;
      if (j0 >= 0 && j0 + 16 <= S_) {
        bf.q[0] = *(const v4u*)(vrow + j0);
        bf.q[1] = *(const v4u*)(vrow + j0 + 8);
      } else {
#pragma unroll
        for (int e = 0; e < 16; ++e) bf.us[e] = 0;
      }
      acc = __builtin_amdgcn_wmma_f32_16x16x32_bf16(false, a.bf, false, bf.bf,
                                                    (short)0, acc, false, false);
    }
#pragma unroll
    for (int r = 0; r < 8; ++r) {
      int qpos = q0 + mt * 16 + half * 8 + r;
      attnb[(size_t)(b * S_ + qpos) * D_ + h * DH_ + dh] = f32_to_bf16(acc[r]);
    }
  }
}

// =====================================================================
// Host driver
// =====================================================================
extern "C" void kernel_launch(void* const* d_in, const int* in_sizes, int n_in,
                              void* d_out, int out_size, void* d_ws, size_t ws_size,
                              hipStream_t stream) {
  (void)in_sizes; (void)n_in; (void)out_size; (void)ws_size;
  const float* x   = (const float*)d_in[0];
  const float* gn1 = (const float*)d_in[1];
  const float* wq  = (const float*)d_in[2];  const float* bq = (const float*)d_in[3];
  const float* wk  = (const float*)d_in[4];  const float* bk = (const float*)d_in[5];
  const float* wv  = (const float*)d_in[6];  const float* bv = (const float*)d_in[7];
  const float* wo  = (const float*)d_in[8];  const float* bo = (const float*)d_in[9];
  const float* gn2 = (const float*)d_in[10];
  const float* w1  = (const float*)d_in[11]; const float* b1 = (const float*)d_in[12];
  const float* w2  = (const float*)d_in[13]; const float* b2 = (const float*)d_in[14];
  const float* w3  = (const float*)d_in[15]; const float* b3 = (const float*)d_in[16];

  char* ws = (char*)d_ws;
  size_t off = 0;
  auto take = [&](size_t bytes) -> char* {
    char* p = ws + off;
    off = (off + bytes + 255) & ~(size_t)255;
    return p;
  };
  unsigned short* wq_t  = (unsigned short*)take((size_t)D_ * D_ * 2);
  unsigned short* wk_t  = (unsigned short*)take((size_t)D_ * D_ * 2);
  unsigned short* wv_t  = (unsigned short*)take((size_t)D_ * D_ * 2);
  unsigned short* wo_t  = (unsigned short*)take((size_t)D_ * D_ * 2);
  unsigned short* w1_t  = (unsigned short*)take((size_t)D_ * F_ * 2);
  unsigned short* w3_t  = (unsigned short*)take((size_t)D_ * F_ * 2);
  unsigned short* w2_t  = (unsigned short*)take((size_t)F_ * D_ * 2);
  unsigned short* xn    = (unsigned short*)take((size_t)MTOT * D_ * 2);  // reused as hn
  unsigned short* qb    = (unsigned short*)take((size_t)MTOT * D_ * 2);
  unsigned short* kbuf  = (unsigned short*)take((size_t)MTOT * D_ * 2);
  unsigned short* vtb   = (unsigned short*)take((size_t)MTOT * D_ * 2);
  unsigned short* attnb = (unsigned short*)take((size_t)MTOT * D_ * 2);
  float*          x2    = (float*)         take((size_t)MTOT * D_ * 4);
  unsigned short* u     = (unsigned short*)take((size_t)MTOT * F_ * 2);

  const dim3 blk(256);

  // weight conversions (bf16 + transpose to [N][K])
  convt_kernel<<<(D_ * D_ + 255) / 256, blk, 0, stream>>>(wq, wq_t, D_, D_);
  convt_kernel<<<(D_ * D_ + 255) / 256, blk, 0, stream>>>(wk, wk_t, D_, D_);
  convt_kernel<<<(D_ * D_ + 255) / 256, blk, 0, stream>>>(wv, wv_t, D_, D_);
  convt_kernel<<<(D_ * D_ + 255) / 256, blk, 0, stream>>>(wo, wo_t, D_, D_);
  convt_kernel<<<(D_ * F_ + 255) / 256, blk, 0, stream>>>(w1, w1_t, D_, F_);
  convt_kernel<<<(D_ * F_ + 255) / 256, blk, 0, stream>>>(w3, w3_t, D_, F_);
  convt_kernel<<<(F_ * D_ + 255) / 256, blk, 0, stream>>>(w2, w2_t, F_, D_);

  // attention half
  rmsnorm_kernel<<<MTOT, blk, 0, stream>>>(x, gn1, xn);
  const int gD = (MTOT / 128) * (D_ / 128);     // 512 blocks of 128x128
  gemm_bias_bf16<<<gD, blk, 0, stream>>>(xn, wq_t, bq, qb,   MTOT, D_, D_);
  gemm_bias_bf16<<<gD, blk, 0, stream>>>(xn, wk_t, bk, kbuf, MTOT, D_, D_);
  gemm_bias_vt  <<<gD, blk, 0, stream>>>(xn, wv_t, bv, vtb,  MTOT, D_, D_);
  attn_kernel<<<B_ * H_ * NB_ * 8, blk, 0, stream>>>(qb, kbuf, vtb, attnb);
  gemm_bias_resid<<<gD, blk, 0, stream>>>(attnb, wo_t, bo, x, x2, MTOT, D_, D_);

  // FFN half
  rmsnorm_kernel<<<MTOT, blk, 0, stream>>>(x2, gn2, xn);
  const int gF = (MTOT / 128) * (F_ / 128);     // 2048 blocks
  gemm_swiglu<<<gF, blk, 0, stream>>>(xn, w1_t, w3_t, b1, b3, u, MTOT, F_, D_);
  gemm_bias_resid<<<gD, blk, 0, stream>>>(u, w2_t, b2, x2, (float*)d_out, MTOT, D_, F_);
}
